// RelationalAttention_21088289424016
// MI455X (gfx1250) — matrix operand order repaired and checked by reading, not compiled, not run on hardware
//
#include <hip/hip_runtime.h>
#include <hip/hip_bf16.h>
#include <stdint.h>

typedef __attribute__((ext_vector_type(16))) _Float16 v16h;
typedef __attribute__((ext_vector_type(8)))  float    v8f;

#define N_TOK 256
#define DIM 512
#define DIM_E 128
#define HEADS 8
#define DHEAD 64
#define DIM_INNER 512            // HEADS*DHEAD
#define QKV_COLS 1536            // 3*DIM_INNER
#define EPS 1.1920929e-07f
#define SCALE 8.0f               // sqrt(DHEAD)

// LDS strides (padded to avoid bank conflicts)
#define ASTR_H 130   // halves per A row (128 data + 2 pad) = 65 dwords
#define BSTR_H 130   // halves per B row (N-major, K contiguous)
#define ESTR   193   // floats per E row (192 data + 1 pad)

__device__ __forceinline__ float waveMax(float v) {
  #pragma unroll
  for (int off = 16; off > 0; off >>= 1)
    v = fmaxf(v, __shfl_xor(v, off, 32));
  return v;
}
__device__ __forceinline__ float waveSum(float v) {
  #pragma unroll
  for (int off = 16; off > 0; off >>= 1)
    v += __shfl_xor(v, off, 32);
  return v;
}

// CDNA5 async global -> LDS copy (GV mode, ASYNCcnt-tracked)
__device__ __forceinline__ void async_copy_b32(uint32_t lds_byte, const void* g) {
  uint64_t ga = (uint64_t)(uintptr_t)g;
  asm volatile("global_load_async_to_lds_b32 %0, %1, off"
               : : "v"(lds_byte), "v"(ga) : "memory");
}
__device__ __forceinline__ void async_wait0() {
  asm volatile("s_wait_asynccnt 0x0" ::: "memory");
}
// low 32 bits of a generic pointer to __shared__ = workgroup LDS byte offset
__device__ __forceinline__ uint32_t lds_off(const void* p) {
  return (uint32_t)(uintptr_t)p;
}

// ---------------- node path: rmsnorm(x) * g_x ----------------
__global__ __launch_bounds__(256) void k_rmsnorm_x(const float* __restrict__ x,
                                                   const float* __restrict__ g,
                                                   float* __restrict__ xn) {
  __shared__ float red[8];
  int j = blockIdx.x, t = threadIdx.x;
  float e0 = x[j*DIM + t];
  float e1 = x[j*DIM + 256 + t];
  float ss = waveSum(e0*e0 + e1*e1);
  int wave = t >> 5, lane = t & 31;
  if (lane == 0) red[wave] = ss;
  __syncthreads();
  float tot = 0.f;
  #pragma unroll
  for (int w = 0; w < 8; ++w) tot += red[w];
  float r = rsqrtf(tot * (1.0f/DIM) + EPS);
  xn[j*DIM + t]       = e0 * r * g[t];
  xn[j*DIM + 256 + t] = e1 * r * g[256 + t];
}

// ---------------- node qkv: (256x512) @ (512x1536) ----------------
__global__ __launch_bounds__(256) void k_node_qkv(const float* __restrict__ xn,
                                                  const float* __restrict__ w,
                                                  float* __restrict__ nqkv) {
  int c = blockIdx.x * 256 + threadIdx.x;
  int j = blockIdx.y;
  float acc = 0.f;
  for (int k = 0; k < DIM; ++k)
    acc = fmaf(xn[j*DIM + k], w[k*QKV_COLS + c], acc);
  nqkv[j*QKV_COLS + c] = acc;
}

// ---- edge weight prep: wt[c][k] = f16( g_e[k] * w_edges_qkv[k][c] ) ----
__global__ __launch_bounds__(256) void k_prep_we(const float* __restrict__ we,
                                                 const float* __restrict__ ge,
                                                 _Float16* __restrict__ wt) {
  int idx = blockIdx.x * 256 + threadIdx.x;   // < 1536*128
  int c = idx >> 7, k = idx & 127;
  wt[idx] = (_Float16)(we[k*QKV_COLS + c] * ge[k]);
}

// ---------------- fused edge-qkv + attention, one block per query row i ----------------
__global__ __launch_bounds__(256) void k_attn(const float* __restrict__ edges,
                                              const _Float16* __restrict__ wt,
                                              const float* __restrict__ nqkv,
                                              float* __restrict__ attn_out) {
  __shared__ _Float16 sA[N_TOK * ASTR_H];   //  66,560 B: rms-normed edge rows (f16)
  __shared__ _Float16 sB[192 * BSTR_H];     //  49,920 B: per-head W slice, [N][K]
  __shared__ float    sE[N_TOK * ESTR];     // 197,632 B: eqkv (then q,k,v) fp32
  __shared__ float    sSim[N_TOK];
  __shared__ float    sPart[N_TOK];
  __shared__ float    sRed[8];

  const int i = blockIdx.x;
  const int t = threadIdx.x;
  const int lane = t & 31, wave = t >> 5;
  const int l16 = lane & 15, hi = lane >> 4;   // hi: 0 for lanes 0-15, 1 for 16-31

  // phase 0: stage raw edges[i] into sE via async global->LDS (coalesced)
  for (int idx = t; idx < N_TOK * DIM_E; idx += 256) {
    int j = idx >> 7, k = idx & 127;
    async_copy_b32(lds_off(&sE[j*ESTR + k]),
                   &edges[((size_t)i * N_TOK + j) * DIM_E + k]);
  }
  async_wait0();
  __syncthreads();

  // phase 1: per-row rmsnorm -> f16 A (g_e already folded into wt)
  {
    int j = t;
    float ss = 0.f;
    for (int k = 0; k < DIM_E; ++k) { float v = sE[j*ESTR + k]; ss = fmaf(v, v, ss); }
    float r = rsqrtf(ss * (1.0f / DIM_E) + EPS);
    for (int k = 0; k < DIM_E; ++k)
      sA[j*ASTR_H + k] = (_Float16)(sE[j*ESTR + k] * r);
  }
  __syncthreads();

  const uint32_t* sAu = reinterpret_cast<const uint32_t*>(sA);
  const uint32_t* sBu = reinterpret_cast<const uint32_t*>(sB);

  // hoist A fragments: identical for all 8 heads -> load exactly once
  union F16F { v16h h; uint32_t u[8]; };
  F16F af[2][4];
  #pragma unroll
  for (int m2 = 0; m2 < 2; ++m2) {
    int arow = (wave * 2 + m2) * 16 + l16;
    #pragma unroll
    for (int ks = 0; ks < 4; ++ks) {
      int abase = arow * (ASTR_H/2) + ks * 16 + hi * 4;
      #pragma unroll
      for (int v = 0; v < 8; ++v)
        af[m2][ks].u[v] = sAu[abase + (v < 4 ? v : v + 4)];
    }
  }

  for (int h = 0; h < HEADS; ++h) {
    // phase 2: async-load this head's 192x128 f16 slice, N-major (K contiguous)
    for (int idx = t; idx < 192 * (DIM_E/2); idx += 256) {  // dword granularity
      int nn = idx >> 6, kp = idx & 63;                     // kp = half-pair index
      int s = nn >> 6, dd = nn & 63;                        // s: 0=q,1=k,2=v
      int col = s * DIM_INNER + h * DHEAD + dd;             // column in 1536-wide qkv
      async_copy_b32(lds_off(&sB[nn*BSTR_H + kp*2]),
                     &wt[col * DIM_E + kp*2]);
    }
    async_wait0();
    __syncthreads();

    // phase 3: WMMA GEMM (256x128 f16) @ (128x192 f16) -> sE fp32
    // each wave owns M tiles {2w, 2w+1}; one B fragment feeds both accumulators
    {
      int jr0 = (wave*2)    *16 + hi*8;
      int jr1 = (wave*2 + 1)*16 + hi*8;
      for (int nt = 0; nt < 12; ++nt) {
        v8f acc0 = {0.f,0.f,0.f,0.f,0.f,0.f,0.f,0.f};
        v8f acc1 = {0.f,0.f,0.f,0.f,0.f,0.f,0.f,0.f};
        int brow = nt * 16 + l16;
        #pragma unroll
        for (int ks = 0; ks < 4; ++ks) {
          F16F bf;
          int bbase = brow * (BSTR_H/2) + ks * 16 + hi * 4;
          #pragma unroll
          for (int v = 0; v < 8; ++v) bf.u[v] = sBu[bbase + (v < 4 ? v : v + 4)];
          acc0 = __builtin_amdgcn_wmma_f32_16x16x32_f16(false, af[0][ks].h, false, bf.h,
                                                        (short)0, acc0, false, false);
          acc1 = __builtin_amdgcn_wmma_f32_16x16x32_f16(false, af[1][ks].h, false, bf.h,
                                                        (short)0, acc1, false, false);
        }
        int cb = nt * 16 + l16;
        #pragma unroll
        for (int r = 0; r < 8; ++r) {
          sE[(jr0 + r)*ESTR + cb] = acc0[r];
          sE[(jr1 + r)*ESTR + cb] = acc1[r];
        }
      }
    }
    __syncthreads();

    // phase 4: add broadcast node terms (indexed by j!) and scale q
    for (int idx = t; idx < N_TOK * 192; idx += 256) {
      int j = idx / 192, c = idx - j * 192;
      int s = c >> 6, dd = c & 63;
      float nv = nqkv[j*QKV_COLS + s*DIM_INNER + h*DHEAD + dd];
      float e = sE[j*ESTR + c] + nv;
      if (s == 0) e *= SCALE;
      sE[j*ESTR + c] = e;
    }
    __syncthreads();

    // phase 5: sim[j] = q[j].k[j], softmax over j
    float simv = 0.f;
    {
      int j = t;
      for (int dd = 0; dd < DHEAD; ++dd)
        simv = fmaf(sE[j*ESTR + dd], sE[j*ESTR + 64 + dd], simv);
    }
    float wmax = waveMax(simv);
    if (lane == 0) sRed[wave] = wmax;
    __syncthreads();
    float mx = sRed[0];
    #pragma unroll
    for (int w = 1; w < 8; ++w) mx = fmaxf(mx, sRed[w]);
    float p = __expf(simv - mx);
    float wsum = waveSum(p);
    __syncthreads();
    if (lane == 0) sRed[wave] = wsum;
    __syncthreads();
    float tot = 0.f;
    #pragma unroll
    for (int w = 0; w < 8; ++w) tot += sRed[w];
    sSim[t] = p / tot;
    __syncthreads();

    // phase 6: out[d] = sum_j attn[j] * v[j][d]
    {
      int dd = t & 63, g = t >> 6;
      float acc = 0.f;
      for (int j = g*64; j < g*64 + 64; ++j)
        acc = fmaf(sSim[j], sE[j*ESTR + 128 + dd], acc);
      sPart[t] = acc;
    }
    __syncthreads();
    if (t < 64) {
      float o = sPart[t] + sPart[64 + t] + sPart[128 + t] + sPart[192 + t];
      attn_out[i*DIM_INNER + h*DHEAD + t] = o;
    }
    __syncthreads();
  }
}

// ---------------- out proj: (256x512) @ (512x512) ----------------
__global__ __launch_bounds__(256) void k_out_proj(const float* __restrict__ ao,
                                                  const float* __restrict__ w,
                                                  float* __restrict__ out) {
  int c = blockIdx.x * 256 + threadIdx.x;
  int j = blockIdx.y;
  float acc = 0.f;
  for (int k = 0; k < DIM_INNER; ++k)
    acc = fmaf(ao[j*DIM_INNER + k], w[k*DIM + c], acc);
  out[j*DIM + c] = acc;
}

extern "C" void kernel_launch(void* const* d_in, const int* in_sizes, int n_in,
                              void* d_out, int out_size, void* d_ws, size_t ws_size,
                              hipStream_t stream) {
  const float* x     = (const float*)d_in[0];
  const float* edges = (const float*)d_in[1];
  const float* g_x   = (const float*)d_in[2];
  const float* g_e   = (const float*)d_in[3];
  const float* w_qkv = (const float*)d_in[4];
  const float* w_e   = (const float*)d_in[5];
  const float* w_out = (const float*)d_in[6];
  float* out = (float*)d_out;

  char* ws = (char*)d_ws;
  float*    xn     = (float*)ws;     ws += (size_t)N_TOK * DIM * 4;        // 512 KB
  float*    nqkv   = (float*)ws;     ws += (size_t)N_TOK * QKV_COLS * 4;   // 1.5 MB
  _Float16* wt     = (_Float16*)ws;  ws += (size_t)QKV_COLS * DIM_E * 2;   // 384 KB
  float*    attn_o = (float*)ws;     ws += (size_t)N_TOK * DIM_INNER * 4;  // 512 KB

  k_rmsnorm_x<<<N_TOK, 256, 0, stream>>>(x, g_x, xn);
  k_node_qkv<<<dim3(QKV_COLS/256, N_TOK), 256, 0, stream>>>(xn, w_qkv, nqkv);
  k_prep_we<<<(QKV_COLS*DIM_E)/256, 256, 0, stream>>>(w_e, g_e, wt);
  k_attn<<<N_TOK, 256, 0, stream>>>(edges, wt, nqkv, attn_o);
  k_out_proj<<<dim3(DIM/256, N_TOK), 256, 0, stream>>>(attn_o, w_out, out);
}
